// DTransformer_10909216932644
// MI455X (gfx1250) — compile-verified, matrix operand order checked by reading, not run on hardware
//
#include <hip/hip_runtime.h>
#include <hip/hip_bf16.h>

typedef __attribute__((ext_vector_type(16))) _Float16 v16h;
typedef __attribute__((ext_vector_type(8)))  _Float16 v8h;
typedef __attribute__((ext_vector_type(4)))  _Float16 v4h;
typedef __attribute__((ext_vector_type(8)))  float    v8f;

#define FLAG_BIAS   1
#define FLAG_RELU   2
#define FLAG_ACC    4
#define FLAG_STOREH 8

// Use GLOBAL_LOAD_ASYNC_TO_LDS_B128 (ASYNCcnt path) for f16 A-tile staging.
#define DT_ASYNC_LDS 1

__device__ __forceinline__ v16h cat8(v8h a, v8h b) {
    v16h r;
#pragma unroll
    for (int i = 0; i < 8; ++i) { r[i] = a[i]; r[8 + i] = b[i]; }
    return r;
}

#if DT_ASYNC_LDS
__device__ __forceinline__ unsigned dt_lds_off(void* p) {
    return (unsigned)(size_t)((__attribute__((address_space(3))) void*)p);
}
#endif

// ---------------------------------------------------------------------------
// Embedding: x[row, d] = toks[row, :] . W_tok[d, :] + W_pos[d, t]
// ---------------------------------------------------------------------------
__global__ __launch_bounds__(256)
void dt_embed_kernel(const float* __restrict__ toks, const float* __restrict__ Wtok,
                     const float* __restrict__ Wpos, float* __restrict__ x,
                     int T, int V, int D, int TMAX) {
    int row = blockIdx.x;            // b*T + t
    int t = row % T;
    __shared__ __align__(16) float tr[64];
    if (threadIdx.x < V) tr[threadIdx.x] = toks[(size_t)row * V + threadIdx.x];
    __syncthreads();
#pragma unroll
    for (int j = 0; j < 4; ++j) {
        int d = threadIdx.x + j * 256;
        const float* w = Wtok + (size_t)d * V;
        float s = 0.f;
#pragma unroll 8
        for (int v = 0; v < 64; ++v) s += tr[v] * w[v];
        x[(size_t)row * D + d] = s + Wpos[(size_t)d * TMAX + t];
    }
}

// ---------------------------------------------------------------------------
// LayerNorm -> f16 output (one block per row of length D=1024)
// ---------------------------------------------------------------------------
__global__ __launch_bounds__(256)
void dt_layernorm_kernel(const float* __restrict__ x, const float* __restrict__ g,
                         const float* __restrict__ b, _Float16* __restrict__ out, int D) {
    const float* row = x + (size_t)blockIdx.x * D;
    float s = 0.f, ss = 0.f;
    for (int i = threadIdx.x; i < D; i += 256) {
        float v = row[i]; s += v; ss += v * v;
    }
#pragma unroll
    for (int off = 16; off >= 1; off >>= 1) {
        s  += __shfl_xor(s,  off);
        ss += __shfl_xor(ss, off);
    }
    __shared__ float rs[8], rss[8];
    int wave = threadIdx.x >> 5, lane = threadIdx.x & 31;
    if (lane == 0) { rs[wave] = s; rss[wave] = ss; }
    __syncthreads();
    if (threadIdx.x == 0) {
        float a = 0.f, c = 0.f;
#pragma unroll
        for (int i = 0; i < 8; ++i) { a += rs[i]; c += rss[i]; }
        rs[0] = a; rss[0] = c;
    }
    __syncthreads();
    float mu  = rs[0] / (float)D;
    float var = rss[0] / (float)D - mu * mu;
    float inv = rsqrtf(var + 1e-5f);
    _Float16* orow = out + (size_t)blockIdx.x * D;
    for (int i = threadIdx.x; i < D; i += 256)
        orow[i] = (_Float16)((row[i] - mu) * inv * g[i] + b[i]);
}

// ---------------------------------------------------------------------------
// WMMA GEMM v2:  out[M,N] = A_f16[M,K] . W_f32[N,K]^T  (+bias)(ReLU)(+=)
// 256 threads = 8 waves; block tile 256x64, K-step 32, double-buffered LDS.
// Wave w owns a 32x64 strip -> 2 A-frags x 4 B-frags = 8 wmma per K-step.
// A tile staged via GLOBAL_LOAD_ASYNC_TO_LDS_B128 (ASYNCcnt), B tile via
// load->v_cvt_pk_f16_f32->ds_store (needs f32->f16 conversion in flight).
// ---------------------------------------------------------------------------
#define G_LDT 40   // LDS row stride in halves (80B -> 16B aligned chunks)
#define BMT   256

__global__ __launch_bounds__(256)
void dt_gemm_wmma_kernel(const _Float16* __restrict__ A, int lda,
                         const float* __restrict__ W,     // [N,K] row-major
                         const float* __restrict__ bias,
                         float* __restrict__ outF, _Float16* __restrict__ outH,
                         int ldo, int Mrows, int N, int K, int flags) {
    __shared__ __align__(16) _Float16 As[2][BMT * G_LDT];
    __shared__ __align__(16) _Float16 Bs[2][64 * G_LDT];

    int tid  = threadIdx.x;
    int lane = tid & 31;
    int wave = tid >> 5;
    int m0 = blockIdx.x * BMT;
    int n0 = blockIdx.y * 64;
    int mW = wave * 32;

    int hi   = lane >> 4;
    int koff = hi * 8;
    int lm   = lane & 15;

    v8f acc[2][4];
#pragma unroll
    for (int a = 0; a < 2; ++a)
#pragma unroll
        for (int c = 0; c < 4; ++c)
#pragma unroll
            for (int r = 0; r < 8; ++r) acc[a][c][r] = 0.f;

    auto stageA = [&](int buf, int k0) {
#pragma unroll
        for (int it = 0; it < 4; ++it) {
            int c = tid + it * 256;               // 1024 chunks of 8 halves
            int r = c >> 2;
            int kp = (c & 3) * 8;
            const _Float16* src = A + (size_t)(m0 + r) * lda + k0 + kp;
            _Float16* dst = &As[buf][r * G_LDT + kp];
#if DT_ASYNC_LDS
            unsigned lo = dt_lds_off(dst);
            asm volatile("global_load_async_to_lds_b128 %0, %1, off"
                         :: "v"(lo), "v"(src) : "memory");
#else
            *(v8h*)dst = *(const v8h*)src;
#endif
        }
    };
    auto stageB = [&](int buf, int k0) {
#pragma unroll
        for (int it = 0; it < 2; ++it) {
            int c = tid + it * 256;               // 512 chunks of 4 floats
            int r = c >> 3;
            int kp = (c & 7) * 4;
            float4 f = *(const float4*)(W + (size_t)(n0 + r) * K + k0 + kp);
            v4h h4;
            h4[0] = (_Float16)f.x; h4[1] = (_Float16)f.y;
            h4[2] = (_Float16)f.z; h4[3] = (_Float16)f.w;
            *(v4h*)&Bs[buf][r * G_LDT + kp] = h4;
        }
    };

    stageA(0, 0);
    stageB(0, 0);
#if DT_ASYNC_LDS
    asm volatile("s_wait_asynccnt 0x0" ::: "memory");
#endif
    __syncthreads();

    int nk = K >> 5;
    for (int ik = 0; ik < nk; ++ik) {
        int buf = ik & 1;
        if (ik + 1 < nk) {                 // prefetch next tile into other buffer
            stageA(buf ^ 1, (ik + 1) * 32);
            stageB(buf ^ 1, (ik + 1) * 32);
        }

        const _Float16* as = &As[buf][0];
        const _Float16* bs = &Bs[buf][0];
        int r0 = (mW + lm) * G_LDT;
        int r1 = (mW + 16 + lm) * G_LDT;
        v16h aF0 = cat8(*(const v8h*)&as[r0 + koff], *(const v8h*)&as[r0 + 16 + koff]);
        v16h aF1 = cat8(*(const v8h*)&as[r1 + koff], *(const v8h*)&as[r1 + 16 + koff]);
#pragma unroll
        for (int c = 0; c < 4; ++c) {
            int nr = (c * 16 + lm) * G_LDT;
            v16h bF = cat8(*(const v8h*)&bs[nr + koff], *(const v8h*)&bs[nr + 16 + koff]);
            acc[0][c] = __builtin_amdgcn_wmma_f32_16x16x32_f16(
                false, aF0, false, bF, (short)0, acc[0][c], false, false);
            acc[1][c] = __builtin_amdgcn_wmma_f32_16x16x32_f16(
                false, aF1, false, bF, (short)0, acc[1][c], false, false);
        }

#if DT_ASYNC_LDS
        asm volatile("s_wait_asynccnt 0x0" ::: "memory");
#endif
        __syncthreads();
    }

    // Epilogue: lane L holds row (ai*16 + r + 8*hi), col (c*16 + lm)
#pragma unroll
    for (int c = 0; c < 4; ++c) {
        int gcol = n0 + c * 16 + lm;
        float bv = (flags & FLAG_BIAS) ? bias[gcol] : 0.f;
#pragma unroll
        for (int ai = 0; ai < 2; ++ai) {
#pragma unroll
            for (int r = 0; r < 8; ++r) {
                int grow = m0 + mW + ai * 16 + r + 8 * hi;
                float v = acc[ai][c][r] + bv;
                if (flags & FLAG_RELU) v = fmaxf(v, 0.f);
                size_t idx = (size_t)grow * ldo + gcol;
                if (flags & FLAG_STOREH)      outH[idx] = (_Float16)v;
                else if (flags & FLAG_ACC)    outF[idx] += v;
                else                          outF[idx] = v;
            }
        }
    }
}

// ---------------------------------------------------------------------------
// Flash-style causal attention over the interleaved QKV f16 buffer.
// qkv row layout per (b,t): [head h][Q 0:64 | K 64:128 | V 128:192].
// Block = (qtile of 64 rows, head, batch); 128 threads = 4 waves x 16 q-rows.
// Accumulates y into hbuf (h += attn(ln(h))).
// ---------------------------------------------------------------------------
__global__ __launch_bounds__(128)
void dt_attn_kernel(const _Float16* __restrict__ qkv, float* __restrict__ hbuf,
                    int B, int T, int H, int C, int D) {
    int b = blockIdx.z, h = blockIdx.y, qt = blockIdx.x;
    int tid = threadIdx.x, lane = tid & 31, wave = tid >> 5;
    int qBase = qt * 64 + wave * 16;
    const int D3 = 3 * H * C;
    const _Float16* base = qkv + (size_t)b * T * D3 + (size_t)h * 3 * C;

    int hi = lane >> 4, koff = hi * 8, lm = lane & 15;

    // Q fragments (scaled by 1/sqrt(C) = 0.125, exact in f16)
    v16h qf[2];
    {
        int qRow = qBase + lm;
        const _Float16* qp = base + (size_t)qRow * D3;
#pragma unroll
        for (int j = 0; j < 2; ++j) {
            v8h q0 = *(const v8h*)(qp + j * 32 + koff);
            v8h q1 = *(const v8h*)(qp + j * 32 + 16 + koff);
            v16h f = cat8(q0, q1);
#pragma unroll
            for (int i = 0; i < 16; ++i) f[i] = f[i] * (_Float16)0.125f;
            qf[j] = f;
        }
    }

    float m_run[8], l_run[8];
    v8f oacc[4];
#pragma unroll
    for (int r = 0; r < 8; ++r) { m_run[r] = -1e30f; l_run[r] = 0.f; }
#pragma unroll
    for (int n = 0; n < 4; ++n)
#pragma unroll
        for (int r = 0; r < 8; ++r) oacc[n][r] = 0.f;

    __shared__ __align__(16) _Float16 Vt[64 * G_LDT];        // V transposed: [c][key]
    __shared__ __align__(16) _Float16 Pl[4][16 * G_LDT];     // per-wave P scratch

    int ktMax = qt * 2 + 1;   // 32-key tiles covering keys 0 .. qt*64+63
    for (int kt = 0; kt <= ktMax; ++kt) {
        int kBase = kt * 32;
        __syncthreads();
        // Stage V tile transposed: 32 keys x 64 c (coalesced global, strided LDS)
        {
            int key  = tid >> 2;              // 0..31
            int cseg = (tid & 3) * 16;        // 0,16,32,48
            const _Float16* vp = base + (size_t)(kBase + key) * D3 + 2 * C + cseg;
            v8h v0 = *(const v8h*)vp;
            v8h v1 = *(const v8h*)(vp + 8);
#pragma unroll
            for (int i = 0; i < 8; ++i) Vt[(cseg + i) * G_LDT + key] = v0[i];
#pragma unroll
            for (int i = 0; i < 8; ++i) Vt[(cseg + 8 + i) * G_LDT + key] = v1[i];
        }
        __syncthreads();

        if (kBase > qBase + 15) continue;     // wave-uniform: tile fully masked

        // ---- S = Q . K^T  (two 16-key col tiles, K-dim = C in two 32-chunks)
        v8f s[2];
#pragma unroll
        for (int t16 = 0; t16 < 2; ++t16) {
#pragma unroll
            for (int r = 0; r < 8; ++r) s[t16][r] = 0.f;
            int keyCol = kBase + t16 * 16 + lm;
            const _Float16* kp = base + (size_t)keyCol * D3 + C;
#pragma unroll
            for (int j = 0; j < 2; ++j) {
                v8h k0 = *(const v8h*)(kp + j * 32 + koff);
                v8h k1 = *(const v8h*)(kp + j * 32 + 16 + koff);
                v16h kF = cat8(k0, k1);
                s[t16] = __builtin_amdgcn_wmma_f32_16x16x32_f16(
                    false, qf[j], false, kF, (short)0, s[t16], false, false);
            }
        }

        // ---- causal mask + online softmax (row lives on 16 lanes of a half)
        float p0[8], p1[8];
#pragma unroll
        for (int r = 0; r < 8; ++r) {
            int q = qBase + r + 8 * hi;
            float s0 = (kBase + lm      <= q) ? s[0][r] : -1e30f;
            float s1 = (kBase + 16 + lm <= q) ? s[1][r] : -1e30f;
            float mx = fmaxf(s0, s1);
#pragma unroll
            for (int off = 1; off < 16; off <<= 1) mx = fmaxf(mx, __shfl_xor(mx, off));
            float mnew  = fmaxf(m_run[r], mx);
            float scale = __expf(m_run[r] - mnew);
            m_run[r] = mnew;
            float e0 = __expf(s0 - mnew), e1 = __expf(s1 - mnew);
            float rs = e0 + e1;
#pragma unroll
            for (int off = 1; off < 16; off <<= 1) rs += __shfl_xor(rs, off);
            l_run[r] = l_run[r] * scale + rs;
            p0[r] = e0; p1[r] = e1;
#pragma unroll
            for (int n = 0; n < 4; ++n) oacc[n][r] *= scale;
        }

        // ---- transpose P through per-wave LDS into an A-fragment
        _Float16* pl = &Pl[wave][0];
#pragma unroll
        for (int r = 0; r < 8; ++r) {
            int rr = r + 8 * hi;
            pl[rr * G_LDT + lm]      = (_Float16)p0[r];
            pl[rr * G_LDT + 16 + lm] = (_Float16)p1[r];
        }
        asm volatile("s_wait_dscnt 0x0" ::: "memory");  // same-wave LDS RAW
        v8h pa0 = *(const v8h*)&pl[lm * G_LDT + koff];
        v8h pa1 = *(const v8h*)&pl[lm * G_LDT + 16 + koff];
        v16h pF = cat8(pa0, pa1);

        // ---- O += P . V   (B-operand K-dim = key index, contiguous in Vt)
#pragma unroll
        for (int n = 0; n < 4; ++n) {
            int c = n * 16 + lm;
            v8h v0 = *(const v8h*)&Vt[c * G_LDT + koff];
            v8h v1 = *(const v8h*)&Vt[c * G_LDT + 16 + koff];
            v16h vF = cat8(v0, v1);
            oacc[n] = __builtin_amdgcn_wmma_f32_16x16x32_f16(
                false, pF, false, vF, (short)0, oacc[n], false, false);
        }
    }

    // Epilogue: h[b,t, h*64 + c] += O / l
#pragma unroll
    for (int n = 0; n < 4; ++n) {
        int d = h * C + n * 16 + lm;
#pragma unroll
        for (int r = 0; r < 8; ++r) {
            int q = qBase + r + 8 * hi;
            hbuf[(size_t)(b * T + q) * D + d] += oacc[n][r] / l_run[r];
        }
    }
}

// ---------------------------------------------------------------------------
// x += h (vectorized)
// ---------------------------------------------------------------------------
__global__ __launch_bounds__(256)
void dt_add_kernel(float* __restrict__ x, const float* __restrict__ h, int n) {
    int i = (blockIdx.x * 256 + threadIdx.x) * 4;
    if (i < n) {
        float4 a = *(float4*)(x + i);
        const float4 b = *(const float4*)(h + i);
        a.x += b.x; a.y += b.y; a.z += b.z; a.w += b.w;
        *(float4*)(x + i) = a;
    }
}

// ---------------------------------------------------------------------------
// Host-side orchestration
// ---------------------------------------------------------------------------
extern "C" void kernel_launch(void* const* d_in, const int* in_sizes, int n_in,
                              void* d_out, int out_size, void* d_ws, size_t ws_size,
                              hipStream_t stream) {
    (void)in_sizes; (void)n_in; (void)out_size; (void)ws_size;
    const float* toks = (const float*)d_in[0];
    const float* Wtok = (const float*)d_in[1];
    const float* Wpos = (const float*)d_in[2];
    const float* Wqkv = (const float*)d_in[3];
    const float* g1   = (const float*)d_in[4];
    const float* be1  = (const float*)d_in[5];
    const float* W1   = (const float*)d_in[6];
    const float* bm1  = (const float*)d_in[7];
    const float* W2   = (const float*)d_in[8];
    const float* bm2  = (const float*)d_in[9];
    const float* g2   = (const float*)d_in[10];
    const float* be2  = (const float*)d_in[11];
    const float* gf   = (const float*)d_in[12];
    const float* bf   = (const float*)d_in[13];
    const float* Wun  = (const float*)d_in[14];
    const float* bun  = (const float*)d_in[15];

    constexpr int Lc = 8, Dc = 1024, Hc = 16, Cc = 64, Mc = 4096, Vc = 64;
    constexpr int Bc = 4, Tc = 2048, TMAXc = 2048;
    const int R = Bc * Tc;                    // 8192 token rows

    char* ws = (char*)d_ws;
    float*     x     = (float*)ws;      ws += (size_t)R * Dc * 4;
    float*     hb    = (float*)ws;      ws += (size_t)R * Dc * 4;
    _Float16*  ln16  = (_Float16*)ws;   ws += (size_t)R * Dc * 2;
    _Float16*  qkv16 = (_Float16*)ws;   ws += (size_t)R * 3 * Dc * 2;
    _Float16*  mid16 = (_Float16*)ws;   ws += (size_t)R * Mc * 2;

    dt_embed_kernel<<<R, 256, 0, stream>>>(toks, Wtok, Wpos, x, Tc, Vc, Dc, TMAXc);

    for (int l = 0; l < Lc; ++l) {
        // h = x
        hipMemcpyAsync(hb, x, (size_t)R * Dc * 4, hipMemcpyDeviceToDevice, stream);
        // ln1(h) -> f16
        dt_layernorm_kernel<<<R, 256, 0, stream>>>(x, g1 + (size_t)l * Dc,
                                                   be1 + (size_t)l * Dc, ln16, Dc);
        // qkv = ln1 @ Wqkv^T   (f16 out)
        dim3 gq(R / BMT, (3 * Dc) / 64);
        dt_gemm_wmma_kernel<<<gq, 256, 0, stream>>>(
            ln16, Dc, Wqkv + (size_t)l * 3 * Dc * Dc, nullptr,
            nullptr, qkv16, 3 * Dc, R, 3 * Dc, Dc, FLAG_STOREH);
        // h += attention(qkv)
        dim3 ga(Tc / 64, Hc, Bc);
        dt_attn_kernel<<<ga, 128, 0, stream>>>(qkv16, hb, Bc, Tc, Hc, Cc, Dc);
        // ln2(h) -> f16
        dt_layernorm_kernel<<<R, 256, 0, stream>>>(hb, g2 + (size_t)l * Dc,
                                                   be2 + (size_t)l * Dc, ln16, Dc);
        // mid = relu(ln2 @ W1^T + bm1)  (f16 out)
        dim3 gm1(R / BMT, Mc / 64);
        dt_gemm_wmma_kernel<<<gm1, 256, 0, stream>>>(
            ln16, Dc, W1 + (size_t)l * Mc * Dc, bm1 + (size_t)l * Mc,
            nullptr, mid16, Mc, R, Mc, Dc, FLAG_STOREH | FLAG_BIAS | FLAG_RELU);
        // h += mid @ W2^T + bm2   (f32 accumulate)
        dim3 gm2(R / BMT, Dc / 64);
        dt_gemm_wmma_kernel<<<gm2, 256, 0, stream>>>(
            mid16, Mc, W2 + (size_t)l * Dc * Mc, bm2 + (size_t)l * Dc,
            hb, nullptr, Dc, R, Dc, Mc, FLAG_BIAS | FLAG_ACC);
        // x += h
        dt_add_kernel<<<(R * Dc / 4 + 255) / 256, 256, 0, stream>>>(x, hb, R * Dc);
    }

    // out = layernorm(x, gf, bf) @ Wun^T + bun
    dt_layernorm_kernel<<<R, 256, 0, stream>>>(x, gf, bf, ln16, Dc);
    dim3 go(R / BMT, Vc / 64);
    dt_gemm_wmma_kernel<<<go, 256, 0, stream>>>(
        ln16, Dc, Wun, bun, (float*)d_out, nullptr, Vc, R, Vc, Dc, FLAG_BIAS);
}
